// VariationalAutoencoder_52518860095620
// MI455X (gfx1250) — matrix operand order repaired and checked by reading, not compiled, not run on hardware
//
#include <hip/hip_runtime.h>
#include <hip/hip_bf16.h>

typedef __attribute__((ext_vector_type(16))) _Float16 v16h;
typedef __attribute__((ext_vector_type(8)))  float    v8f;
typedef __attribute__((ext_vector_type(4)))  int      v4i;

#define AS1 __attribute__((address_space(1)))
#define AS3 __attribute__((address_space(3)))

#if defined(__has_builtin)
#if __has_builtin(__builtin_amdgcn_global_load_async_to_lds_b128) && \
    __has_builtin(__builtin_amdgcn_s_wait_asynccnt)
#define USE_ASYNC 1
#endif
#endif
#ifndef USE_ASYNC
#define USE_ASYNC 0
#endif

#define LEAKY(x) ((x) >= 0.0f ? (x) : 0.01f * (x))
static __device__ __forceinline__ float sigm(float x) { return 1.0f / (1.0f + __expf(-x)); }

// ---------------------------------------------------------------- encoder convs
// 4x4 conv, stride 2, pad 1, + bias + leaky. One thread per output element.
__global__ void conv_leaky(const float* __restrict__ x, const float* __restrict__ w,
                           const float* __restrict__ bias, float* __restrict__ y,
                           int Cin, int Hin, int Win, int Cout, int Hout, int Wout, int total) {
  int idx = blockIdx.x * blockDim.x + threadIdx.x;
  if (idx >= total) return;
  int ow = idx % Wout; int t = idx / Wout;
  int oh = t % Hout;   t /= Hout;
  int co = t % Cout;   int b = t / Cout;
  float acc = bias[co];
  const int HW = Hin * Win;
  for (int kh = 0; kh < 4; ++kh) {
    int ih = oh * 2 - 1 + kh;
    if (ih < 0 || ih >= Hin) continue;
    for (int kw = 0; kw < 4; ++kw) {
      int iw = ow * 2 - 1 + kw;
      if (iw < 0 || iw >= Win) continue;
      const float* xp = x + ((size_t)b * Cin) * HW + (size_t)ih * Win + iw;
      const float* wp = w + (((size_t)co * Cin) * 4 + kh) * 4 + kw;
      for (int ci = 0; ci < Cin; ++ci)
        acc += xp[(size_t)ci * HW] * wp[(size_t)ci * 16];
    }
  }
  y[idx] = LEAKY(acc);
}

// per-channel mean / invstd over (B,H,W). One block (256 thr) per channel.
__global__ void bn_stats(const float* __restrict__ h, float* __restrict__ stats,
                         int C, int HW) {
  const int B = 4;
  int c = blockIdx.x;
  __shared__ float s1[256], s2[256];
  float sum = 0.f, sq = 0.f;
  int n = B * HW;
  for (int i = threadIdx.x; i < n; i += 256) {
    int b = i / HW, r = i % HW;
    float v = h[((size_t)b * C + c) * HW + r];
    sum += v; sq += v * v;
  }
  s1[threadIdx.x] = sum; s2[threadIdx.x] = sq;
  __syncthreads();
  for (int off = 128; off > 0; off >>= 1) {
    if (threadIdx.x < off) { s1[threadIdx.x] += s1[threadIdx.x + off];
                             s2[threadIdx.x] += s2[threadIdx.x + off]; }
    __syncthreads();
  }
  if (threadIdx.x == 0) {
    float m = s1[0] / n;
    float v = s2[0] / n - m * m;
    stats[c] = m;
    stats[C + c] = rsqrtf(v + 1e-5f);
  }
}

__global__ void bn_apply(float* __restrict__ h, const float* __restrict__ stats,
                         int C, int HW, int total) {
  int idx = blockIdx.x * blockDim.x + threadIdx.x;
  if (idx >= total) return;
  int c = (idx / HW) % C;
  h[idx] = (h[idx] - stats[c]) * stats[C + c];
}

// ---------------------------------------------------------------- FC + relu
__global__ void fc_relu(const float* __restrict__ h, const float* __restrict__ fcw,
                        const float* __restrict__ fcb, float* __restrict__ eig) {
  int j = blockIdx.x;
  __shared__ float s[4][256];
  float acc0 = 0.f, acc1 = 0.f, acc2 = 0.f, acc3 = 0.f;
  const float* wrow = fcw + (size_t)j * 25600;
  for (int i = threadIdx.x; i < 25600; i += 256) {
    float wv = wrow[i];
    acc0 += h[i]         * wv;
    acc1 += h[25600 + i] * wv;
    acc2 += h[51200 + i] * wv;
    acc3 += h[76800 + i] * wv;
  }
  s[0][threadIdx.x] = acc0; s[1][threadIdx.x] = acc1;
  s[2][threadIdx.x] = acc2; s[3][threadIdx.x] = acc3;
  __syncthreads();
  for (int off = 128; off > 0; off >>= 1) {
    if (threadIdx.x < off)
      for (int b = 0; b < 4; ++b) s[b][threadIdx.x] += s[b][threadIdx.x + off];
    __syncthreads();
  }
  if (threadIdx.x < 4) {
    float v = s[threadIdx.x][0] + fcb[j];
    eig[(size_t)threadIdx.x * 2025 + j] = v > 0.f ? v : 0.f;
  }
}

// ---------------------------------------------------------------- WMMA fragment packing
// DFT = kron(F,F): entry (j,k) = exp(-2*pi*i*((j1*k1 + j2*k2) mod 45)/45)/45.
static __device__ __forceinline__ void dft_ri(int j, int k, float* r, float* i) {
  int t = ((j / 45) * (k / 45) + (j % 45) * (k % 45)) % 45;
  float ang = -6.283185307179586f * (float)t * (1.0f / 45.0f);
  float s, c;
  __sincosf(ang, &s, &c);
  *r = c * (1.0f / 45.0f);
  *i = s * (1.0f / 45.0f);
}

// A = Dh rows (M=1012 pad 1024, K=2025 pad 2048). Fragment order: [kb][mt] so a
// block's 8-tile slice at fixed kb is contiguous (8 KB) for async copies.
// 16x32 f16 A layout: lane L: m = (L&15), ksec = L>>4; half h ->
//   K = (h<8 ? ksec*8+h : 16+ksec*8+h-8)
__global__ void pack_A(_Float16* __restrict__ AR, _Float16* __restrict__ AI, int total) {
  int idx = blockIdx.x * blockDim.x + threadIdx.x;
  if (idx >= total) return;
  int h    = idx & 15;
  int lane = (idx >> 4) & 31;
  int frag = idx >> 9;          // kb*64 + mt
  int mt = frag & 63, kb = frag >> 6;
  int m    = mt * 16 + (lane & 15);
  int ksec = lane >> 4;
  int kin  = (h < 8) ? (ksec * 8 + h) : (16 + ksec * 8 + (h - 8));
  int k    = kb * 32 + kin;
  float vr = 0.f, vi = 0.f;
  if (m < 1012 && k < 2025) dft_ri(m, k, &vr, &vi);
  AR[idx] = (_Float16)vr;
  AI[idx] = (_Float16)vi;
}

// B[m][n] = e[b][m] * {R,I}[n][m]. Fragment order per batch: [kb][nt].
// 32x16 f16 B layout: lane L: n = (L&15), ksec = (L>>4)&1; half h -> k_local = (h<8 ? ksec*8+h : 16+ksec*8+h-8)
__global__ void pack_B(const float* __restrict__ eig,
                       _Float16* __restrict__ BR, _Float16* __restrict__ BI, int total) {
  int idx = blockIdx.x * blockDim.x + threadIdx.x;
  if (idx >= total) return;
  int h    = idx & 15;
  int lane = (idx >> 4) & 31;
  int frag = (idx >> 9) & 4095; // kb*64 + nt
  int b    = idx >> 21;
  int nt = frag & 63, kb = frag >> 6;
  int n    = nt * 16 + (lane & 15);
  int k    = kb * 32 + ((h < 8) ? (((lane >> 4) & 1) * 8 + h) : (16 + ((lane >> 4) & 1) * 8 + (h - 8)));
  float vr = 0.f, vi = 0.f;
  if (k < 2025 && n < 1012) {
    float e = eig[(size_t)b * 2025 + k];
    dft_ri(n, k, &vr, &vi);
    vr *= e; vi *= e;
  }
  BR[idx] = (_Float16)vr;
  BI[idx] = (_Float16)vi;
}

// ---------------------------------------------------------------- cov GEMM (WMMA + async LDS)
// cov[b] = A_R * B_R[b] + A_I * B_I[b].
// Block = 256 thr (8 waves, 2x4 wave grid) -> 128x128 output block.
// Per K-step: stage 32 KB (8 A-tiles + 8 B-tiles, R & I) global->LDS
// (async, double buffered), each wave does 4x2 register tiles = 16 WMMAs.
__global__ __launch_bounds__(256) void cov_wmma(
    const _Float16* __restrict__ ARh, const _Float16* __restrict__ AIh,
    const _Float16* __restrict__ BRh, const _Float16* __restrict__ BIh,
    float* __restrict__ cov) {
  __shared__ v16h smem[2][1024];   // 2 x 32 KB: [AR(8 frag)|AI(8)|BR(8)|BI(8)] x 512 halfs

  const int tid  = threadIdx.x;
  const int lane = tid & 31;
  const int wave = tid >> 5;
  const int mw = wave >> 2;        // 0..1
  const int nw = wave & 3;         // 0..3

  int blk = blockIdx.x;            // b*64 + mb*8 + nb
  const int nb = blk & 7;  blk >>= 3;
  const int mb = blk & 7;  const int b = blk >> 3;
  const int mt0 = mb * 8;
  const int nt0 = nb * 8;

  const _Float16* srcs[4];
  srcs[0] = ARh + (size_t)mt0 * 512;
  srcs[1] = AIh + (size_t)mt0 * 512;
  srcs[2] = BRh + ((size_t)b * 4096 + nt0) * 512;
  srcs[3] = BIh + ((size_t)b * 4096 + nt0) * 512;

  auto issue_copy = [&](int kb, int buf) {
    const size_t kofs = (size_t)kb * (64 * 512);
#pragma unroll
    for (int jj = 0; jj < 8; ++jj) {
      const int region = jj >> 1;                 // 0:AR 1:AI 2:BR 3:BI
      const int off = ((jj & 1) << 8) + tid;      // 0..511 (16B chunks)
      const _Float16* src = srcs[region] + kofs + (size_t)off * 8;
      _Float16* dst = (_Float16*)smem[buf] + (size_t)region * 4096 + (size_t)off * 8;
#if USE_ASYNC
      __builtin_amdgcn_global_load_async_to_lds_b128(
          (AS1 v4i*)(AS1 void*)src, (AS3 v4i*)(AS3 void*)dst, 0, 0);
#else
      *(float4*)dst = *(const float4*)src;
#endif
    }
  };

  v8f acc[4][2] = {};

  issue_copy(0, 0);
  for (int kb = 0; kb < 64; ++kb) {
    const int cur = kb & 1;
#if USE_ASYNC
    __builtin_amdgcn_s_wait_asynccnt(0);
#endif
    __syncthreads();
    if (kb + 1 < 64) issue_copy(kb + 1, cur ^ 1);

    const v16h* S = smem[cur];
    v16h aR[4], aI[4], bR[2], bI[2];
#pragma unroll
    for (int i = 0; i < 4; ++i) {
      aR[i] = S[       (mw * 4 + i) * 32 + lane];
      aI[i] = S[256 +  (mw * 4 + i) * 32 + lane];
    }
#pragma unroll
    for (int j = 0; j < 2; ++j) {
      bR[j] = S[512 + (nw * 2 + j) * 32 + lane];
      bI[j] = S[768 + (nw * 2 + j) * 32 + lane];
    }
#pragma unroll
    for (int i = 0; i < 4; ++i)
#pragma unroll
      for (int j = 0; j < 2; ++j) {
        acc[i][j] = __builtin_amdgcn_wmma_f32_16x16x32_f16(
            false, aR[i], false, bR[j], (short)0, acc[i][j], false, false);
        acc[i][j] = __builtin_amdgcn_wmma_f32_16x16x32_f16(
            false, aI[i], false, bI[j], (short)0, acc[i][j], false, false);
      }
  }

  // C/D layout: VGPR r -> M = r + 8*(lane>>4); N = lane & 15
  const int ncol = lane & 15;
  const int mhi  = (lane >> 4) << 3;
#pragma unroll
  for (int i = 0; i < 4; ++i) {
    const int mt = mt0 + mw * 4 + i;
#pragma unroll
    for (int j = 0; j < 2; ++j) {
      const int nt = nt0 + nw * 2 + j;
      const int n = nt * 16 + ncol;
      const int mbase = mt * 16 + mhi;
      if (n < 1012) {
#pragma unroll
        for (int r = 0; r < 8; ++r) {
          const int row = mbase + r;
          if (row < 1012) cov[((size_t)b * 1012 + row) * 1012 + n] = acc[i][j][r];
        }
      }
    }
  }
}

// ---------------------------------------------------------------- l = DFT @ (e^2 * zeta)
__global__ void l_dft(const float* __restrict__ eig, const float* __restrict__ epsa,
                      const float* __restrict__ epsb, float* __restrict__ yy) {
  int n = blockIdx.x % 2025;
  int b = blockIdx.x / 2025;
  int n1 = n / 45, n2 = n % 45;
  __shared__ float sr[128], si[128];
  float ar = 0.f, ai = 0.f;
  for (int m = threadIdx.x; m < 2025; m += 128) {
    int t = (n1 * (m / 45) + n2 * (m % 45)) % 45;
    float ang = -6.283185307179586f * (float)t * (1.0f / 45.0f);
    float s, c;
    __sincosf(ang, &s, &c);
    c *= (1.0f / 45.0f); s *= (1.0f / 45.0f);
    float e = eig[(size_t)b * 2025 + m];
    float e2 = e * e;
    float sa = e2 * epsa[(size_t)b * 2025 + m];
    float sb = e2 * epsb[(size_t)b * 2025 + m];
    ar += c * sa - s * sb;
    ai += c * sb + s * sa;
  }
  sr[threadIdx.x] = ar; si[threadIdx.x] = ai;
  __syncthreads();
  for (int off = 64; off > 0; off >>= 1) {
    if (threadIdx.x < off) { sr[threadIdx.x] += sr[threadIdx.x + off];
                             si[threadIdx.x] += si[threadIdx.x + off]; }
    __syncthreads();
  }
  if (threadIdx.x == 0) {
    yy[(size_t)b * 4050 + n]        = sr[0];
    yy[(size_t)b * 4050 + 2025 + n] = si[0];
  }
}

// ---------------------------------------------------------------- decoder
__global__ void deconv1(const float* __restrict__ yy, const float* __restrict__ dw1,
                        const float* __restrict__ db1, float* __restrict__ r, int total) {
  int idx = blockIdx.x * blockDim.x + threadIdx.x;
  if (idx >= total) return;
  int q = idx % 89; int t = idx / 89;
  int p = t % 89;   t /= 89;
  int o = t & 1;    int b = t >> 1;
  float acc = db1[o];
  for (int c = 0; c < 2; ++c)
    for (int kh = 0; kh < 3; ++kh) {
      int ph = p + 1 - kh;
      if (ph < 0 || (ph & 1)) continue;
      int i = ph >> 1;
      if (i >= 45) continue;
      for (int kw = 0; kw < 3; ++kw) {
        int qw = q + 1 - kw;
        if (qw < 0 || (qw & 1)) continue;
        int jj = qw >> 1;
        if (jj >= 45) continue;
        acc += yy[((size_t)(b * 2 + c) * 45 + i) * 45 + jj] *
               dw1[((c * 2 + o) * 3 + kh) * 3 + kw];
      }
    }
  r[idx] = LEAKY(acc);
}

__global__ void deconv2(const float* __restrict__ r, const float* __restrict__ dw2,
                        const float* __restrict__ db2, float* __restrict__ out, int total) {
  int idx = blockIdx.x * blockDim.x + threadIdx.x;
  if (idx >= total) return;
  int q = idx % 91; int t = idx / 91;
  int p = t % 91;   int b = t / 91;
  float acc = db2[0];
  for (int c = 0; c < 2; ++c)
    for (int kh = 0; kh < 3; ++kh) {
      int i = p - kh;
      if (i < 0 || i >= 89) continue;
      for (int kw = 0; kw < 3; ++kw) {
        int j = q - kw;
        if (j < 0 || j >= 89) continue;
        acc += r[((size_t)(b * 2 + c) * 89 + i) * 89 + j] * dw2[(c * 3 + kh) * 3 + kw];
      }
    }
  out[idx] = sigm(acc);
}

// ---------------------------------------------------------------- launch
extern "C" void kernel_launch(void* const* d_in, const int* in_sizes, int n_in,
                              void* d_out, int out_size, void* d_ws, size_t ws_size,
                              hipStream_t stream) {
  const float* x    = (const float*)d_in[0];
  const float* epsa = (const float*)d_in[1];
  const float* epsb = (const float*)d_in[2];
  const float* w1 = (const float*)d_in[3];  const float* b1 = (const float*)d_in[4];
  const float* w2 = (const float*)d_in[5];  const float* b2 = (const float*)d_in[6];
  const float* w3 = (const float*)d_in[7];  const float* b3 = (const float*)d_in[8];
  const float* w4 = (const float*)d_in[9];  const float* b4 = (const float*)d_in[10];
  const float* fcw = (const float*)d_in[11]; const float* fcb = (const float*)d_in[12];
  const float* dw1 = (const float*)d_in[13]; const float* db1 = (const float*)d_in[14];
  const float* dw2 = (const float*)d_in[15]; const float* db2 = (const float*)d_in[16];

  float* outF   = (float*)d_out;
  float* xrec   = outF;                         // 4*1*91*91   = 33124
  float* covOut = outF + 33124;                 // 4*1012*1012 = 4096576
  float* yyOut  = outF + 33124 + 4096576;       // 4*2*45*45   = 16200

  char* ws = (char*)d_ws;
  size_t off = 0;
  auto alloc = [&](size_t bytes) { size_t cur = off; off = (off + bytes + 255) & ~(size_t)255; return cur; };
  float*    h1    = (float*)(ws + alloc(409600u * 4));
  float*    h2    = (float*)(ws + alloc(409600u * 4));
  float*    h3    = (float*)(ws + alloc(204800u * 4));
  float*    h4    = (float*)(ws + alloc(102400u * 4));
  float*    stats = (float*)(ws + alloc(2048u * 4));
  float*    eig   = (float*)(ws + alloc(8100u * 4));
  float*    rbuf  = (float*)(ws + alloc(63368u * 4));
  _Float16* AR    = (_Float16*)(ws + alloc(2097152u * 2));
  _Float16* AI    = (_Float16*)(ws + alloc(2097152u * 2));
  _Float16* BR    = (_Float16*)(ws + alloc(8388608u * 2));
  _Float16* BI    = (_Float16*)(ws + alloc(8388608u * 2));

  const int T = 256;

  pack_A<<<2097152 / T, T, 0, stream>>>(AR, AI, 2097152);

  conv_leaky<<<(409600 + T - 1) / T, T, 0, stream>>>(x,  w1, b1, h1,   1, 80, 80,   64, 40, 40, 409600);
  bn_stats<<<64,  T, 0, stream>>>(h1, stats, 64, 1600);
  bn_apply<<<(409600 + T - 1) / T, T, 0, stream>>>(h1, stats, 64, 1600, 409600);

  conv_leaky<<<(409600 + T - 1) / T, T, 0, stream>>>(h1, w2, b2, h2,  64, 40, 40,  256, 20, 20, 409600);
  bn_stats<<<256, T, 0, stream>>>(h2, stats, 256, 400);
  bn_apply<<<(409600 + T - 1) / T, T, 0, stream>>>(h2, stats, 256, 400, 409600);

  conv_leaky<<<(204800 + T - 1) / T, T, 0, stream>>>(h2, w3, b3, h3, 256, 20, 20,  512, 10, 10, 204800);
  bn_stats<<<512, T, 0, stream>>>(h3, stats, 512, 100);
  bn_apply<<<(204800 + T - 1) / T, T, 0, stream>>>(h3, stats, 512, 100, 204800);

  conv_leaky<<<(102400 + T - 1) / T, T, 0, stream>>>(h3, w4, b4, h4, 512, 10, 10, 1024,  5,  5, 102400);
  bn_stats<<<1024, T, 0, stream>>>(h4, stats, 1024, 25);
  bn_apply<<<(102400 + T - 1) / T, T, 0, stream>>>(h4, stats, 1024, 25, 102400);

  fc_relu<<<2025, T, 0, stream>>>(h4, fcw, fcb, eig);

  pack_B<<<8388608 / T, T, 0, stream>>>(eig, BR, BI, 8388608);

  // cov: 4 batches * 8x8 blocks of 128x128
  cov_wmma<<<256, T, 0, stream>>>(AR, AI, BR, BI, covOut);

  l_dft<<<4 * 2025, 128, 0, stream>>>(eig, epsa, epsb, yyOut);

  deconv1<<<(63368 + T - 1) / T, T, 0, stream>>>(yyOut, dw1, db1, rbuf, 63368);
  deconv2<<<(33124 + T - 1) / T, T, 0, stream>>>(rbuf, dw2, db2, xrec, 33124);
}
